// RnnEncoder_13039520711439
// MI455X (gfx1250) — matrix operand order repaired and checked by reading, not compiled, not run on hardware
//
#include <hip/hip_runtime.h>
#include <hip/hip_bf16.h>
#include <math.h>

// ---------------------------------------------------------------------------
// RnnEncoder for MI455X (gfx1250, wave32, WMMA 16x16x32 bf16)
//
// Pipeline:
//   1) cvt: x -> bf16 ; W_mlp0/W_ih/W_hh -> bf16, transposed to N-major
//   2) WMMA GEMM: m = relu(x @ W_mlp0 + b)           (f32 out)
//   3) scores:   e = exp(tanh(m@W1+b1)@W2 + b2)      (tiny N, VALU)
//   4) zinv:     running prefix sums of e -> 1/Z     (B*R tiny scans)
//   5) scan:     M = (1/R) sum_r zinv * prefix(e*m)  (linear-time causal attn)
//   6) WMMA GEMM: gi = M @ W_ih + b_i
//   7) GRU: persistent single-WGP kernel, h in LDS (f32 + bf16 frag form),
//           32 waves x 3 gate tiles x v_wmma_f32_16x16x32_bf16 per k-step.
// ---------------------------------------------------------------------------

#define BB   16
#define SS   512
#define IN_  128
#define DD   512
#define HH   512
#define DA_  30
#define RR   10
#define H3   1536
#define MSROWS (BB * SS)

typedef __attribute__((ext_vector_type(16))) __bf16 v16bf;
typedef __attribute__((ext_vector_type(8)))  __bf16 v8bf;
typedef __attribute__((ext_vector_type(8)))  float  v8f;

// ---- WMMA fragment loaders -------------------------------------------------
// A (MxK, 16x32, row-major, leading dim ld):
//   lane L: row M = L&15 ; element e -> K = (L>>4)*16 + e
__device__ __forceinline__ v16bf load_A_frag(const __bf16* tile, int ld, int lane) {
  const int half = lane >> 4;
  const __bf16* p = tile + (lane & 15) * ld + (half << 4);
  return *(const v16bf*)(p);
}

// B (KxN, 32x16) supplied pre-transposed N-major (N rows x K cols, ld=K):
//   lane L: col N = L&15 ; element e -> K = (L>>4)*16 + e  (16 contiguous bf16)
__device__ __forceinline__ v16bf load_B_fragT(const __bf16* tileT, int ld, int lane) {
  const __bf16* p = tileT + (lane & 15) * ld + ((lane >> 4) << 4);
  return *(const v16bf*)(p);
}

// ---- conversion kernels ----------------------------------------------------
__global__ void k_cvt_bf16(const float* __restrict__ src, __bf16* __restrict__ dst, int n) {
  int i = blockIdx.x * blockDim.x + threadIdx.x;
  if (i < n) dst[i] = (__bf16)src[i];
}

// src: K x N row-major  ->  dst: N x K (N-major) bf16
__global__ void k_cvt_bf16_T(const float* __restrict__ src, __bf16* __restrict__ dst,
                             int K, int N) {
  int i = blockIdx.x * blockDim.x + threadIdx.x;
  if (i < K * N) {
    int k = i / N, n = i % N;
    dst[n * K + k] = (__bf16)src[i];
  }
}

// ---- WMMA GEMM: Cf[M,N] = opt_relu(A_bf16[M,K] @ B^T(N-major)[N,K] + bias) -
// one wave = one 16(M) x 64(N) strip: A fragment reused across 4 B tiles
__global__ void k_gemm_wmma(const __bf16* __restrict__ A, const __bf16* __restrict__ BT,
                            const float* __restrict__ bias, float* __restrict__ Cf,
                            int Mrows, int N, int K, int do_relu) {
  const int lane = threadIdx.x & 31;
  const int gw = blockIdx.x * (blockDim.x >> 5) + (threadIdx.x >> 5);
  const int nstrips = N >> 6;
  const int strips = (Mrows >> 4) * nstrips;
  if (gw >= strips) return;                 // wave-uniform: EXEC all-ones for WMMA
  const int m0 = (gw / nstrips) << 4;
  const int n0 = (gw % nstrips) << 6;

  v8f acc[4] = {};
  for (int k0 = 0; k0 < K; k0 += 32) {
    v16bf a = load_A_frag(A + m0 * K + k0, K, lane);
    __builtin_prefetch(A + m0 * K + k0 + 64, 0, 1);   // global_prefetch_b8
#pragma unroll
    for (int j = 0; j < 4; ++j) {
      v16bf b = load_B_fragT(BT + (n0 + j * 16) * K + k0, K, lane);
      acc[j] = __builtin_amdgcn_wmma_f32_16x16x32_bf16(
          false, a, false, b, (short)0, acc[j], false, false);
    }
  }
  // C layout: VGPR i, lanes 0-15: M=i ; lanes 16-31: M=8+i ; N = lane&15
  const int half = lane >> 4;
  const int nn = lane & 15;
#pragma unroll
  for (int j = 0; j < 4; ++j) {
    int col = n0 + j * 16 + nn;
    float bv = bias ? bias[col] : 0.f;
#pragma unroll
    for (int i = 0; i < 8; ++i) {
      int row = m0 + i + (half << 3);
      float v = acc[j][i] + bv;
      if (do_relu) v = fmaxf(v, 0.f);
      Cf[row * N + col] = v;
    }
  }
}

// ---- attention scores: e = exp(tanh(m@W1+b1)@W2+b2), one block per row ----
__global__ void k_scores(const float* __restrict__ m, const float* __restrict__ W1,
                         const float* __restrict__ b1, const float* __restrict__ W2,
                         const float* __restrict__ b2, float* __restrict__ e_out) {
  __shared__ float row[DD];
  __shared__ float tact[DA_];
  const int rid = blockIdx.x;                        // b*S + k
  for (int j = threadIdx.x; j < DD; j += blockDim.x) row[j] = m[rid * DD + j];
  __syncthreads();
  if (threadIdx.x < DA_) {
    float acc = b1[threadIdx.x];
    for (int j = 0; j < DD; ++j) acc = fmaf(row[j], W1[j * DA_ + threadIdx.x], acc);
    tact[threadIdx.x] = tanhf(acc);
  }
  __syncthreads();
  if (threadIdx.x < RR) {
    float acc = b2[threadIdx.x];
#pragma unroll
    for (int j = 0; j < DA_; ++j) acc = fmaf(tact[j], W2[j * RR + threadIdx.x], acc);
    e_out[rid * RR + threadIdx.x] = __expf(acc);
  }
}

// ---- prefix-sum of e over k -> 1/Z[b,t,r] (tiny: B*R scans of length S) ----
__global__ void k_zinv(const float* __restrict__ e, float* __restrict__ zinv) {
  int t = blockIdx.x * blockDim.x + threadIdx.x;
  if (t >= BB * RR) return;
  int b = t / RR, r = t % RR;
  float Z = 0.f;
  for (int k = 0; k < SS; ++k) {
    int idx = (b * SS + k) * RR + r;
    Z += e[idx];
    zinv[idx] = 1.f / Z;
  }
}

// ---- linear-time causal attention: M[b,t,d] = (1/R) sum_r zinv*prefix(e*m) -
__global__ void k_scan(const float* __restrict__ m, const float* __restrict__ e,
                       const float* __restrict__ zinv, __bf16* __restrict__ Mbf) {
  const int b = blockIdx.x >> 2;
  const int d = ((blockIdx.x & 3) << 7) + threadIdx.x;   // blockDim = 128
  float C[RR];
#pragma unroll
  for (int r = 0; r < RR; ++r) C[r] = 0.f;
  const float invR = 1.0f / (float)RR;
  for (int k = 0; k < SS; ++k) {
    const int rb = b * SS + k;
    float mv = m[rb * DD + d];
    float acc = 0.f;
#pragma unroll
    for (int r = 0; r < RR; ++r) {
      C[r] = fmaf(e[rb * RR + r], mv, C[r]);
      acc = fmaf(zinv[rb * RR + r], C[r], acc);
    }
    Mbf[rb * DD + d] = (__bf16)(acc * invR);
  }
}

// ---- persistent GRU: 1 block, 32 waves, h in LDS, W_hh bf16 L2-resident ----
__global__ void __launch_bounds__(1024, 1)
k_gru(const float* __restrict__ gi, const __bf16* __restrict__ WhhT,
      const float* __restrict__ b_h, float* __restrict__ out) {
  __shared__ __bf16 h_bf[BB * HH];   // A-fragment source (ds_load_b128)
  __shared__ float  h_f[BB * HH];    // f32 master copy of h
  const int tid = threadIdx.x;
  const int lane = tid & 31;
  const int w = tid >> 5;                       // 0..31: owns cols [16w,16w+16)
  for (int j = tid; j < BB * HH; j += blockDim.x) { h_f[j] = 0.f; h_bf[j] = (__bf16)0.f; }
  __syncthreads();
  const int half = lane >> 4;
  const int nn = lane & 15;
  const int col0 = w << 4;

  for (int t = 0; t < SS; ++t) {
    v8f accR = {}, accZ = {}, accN = {};
    for (int kb = 0; kb < HH; kb += 32) {
      v16bf a = load_A_frag(&h_bf[kb], HH, lane);        // shared A across gates
      v16bf br = load_B_fragT(WhhT + (0 * HH + col0) * HH + kb, HH, lane);
      v16bf bz = load_B_fragT(WhhT + (1 * HH + col0) * HH + kb, HH, lane);
      v16bf bn = load_B_fragT(WhhT + (2 * HH + col0) * HH + kb, HH, lane);
      accR = __builtin_amdgcn_wmma_f32_16x16x32_bf16(false, a, false, br, (short)0, accR, false, false);
      accZ = __builtin_amdgcn_wmma_f32_16x16x32_bf16(false, a, false, bz, (short)0, accZ, false, false);
      accN = __builtin_amdgcn_wmma_f32_16x16x32_bf16(false, a, false, bn, (short)0, accN, false, false);
    }
    __syncthreads();                  // all waves done reading h for step t
    const int j = col0 + nn;
#pragma unroll
    for (int i = 0; i < 8; ++i) {
      const int b = i + (half << 3);                     // batch row 0..15
      const float* girow = gi + ((size_t)(b * SS + t)) * H3;
      float hr = accR[i] + b_h[j];
      float hz = accZ[i] + b_h[HH + j];
      float hn = accN[i] + b_h[2 * HH + j];
      float rt = 1.f / (1.f + __expf(-(girow[j]          + hr)));
      float zt = 1.f / (1.f + __expf(-(girow[HH + j]     + hz)));
      float nt = tanhf(girow[2 * HH + j] + rt * hn);
      float hold = h_f[b * HH + j];
      float hnew = nt + zt * (hold - nt);
      h_f[b * HH + j] = hnew;
      h_bf[b * HH + j] = (__bf16)hnew;
      out[((size_t)(b * SS + t)) * HH + j] = hnew;
    }
    __syncthreads();                  // h fully updated before next step
  }
}

// ---------------------------------------------------------------------------
extern "C" void kernel_launch(void* const* d_in, const int* in_sizes, int n_in,
                              void* d_out, int out_size, void* d_ws, size_t ws_size,
                              hipStream_t stream) {
  (void)in_sizes; (void)n_in; (void)out_size; (void)ws_size;
  const float* x      = (const float*)d_in[0];
  const float* W_mlp0 = (const float*)d_in[1];
  const float* b_mlp0 = (const float*)d_in[2];
  const float* W1     = (const float*)d_in[3];
  const float* b1     = (const float*)d_in[4];
  const float* W2     = (const float*)d_in[5];
  const float* b2     = (const float*)d_in[6];
  const float* W_ih   = (const float*)d_in[7];
  const float* W_hh   = (const float*)d_in[8];
  const float* b_i    = (const float*)d_in[9];
  const float* b_h    = (const float*)d_in[10];
  float* out = (float*)d_out;

  char* ws = (char*)d_ws;
  size_t off = 0;
  auto wsalloc = [&](size_t bytes) -> char* {
    char* p = ws + off;
    off += (bytes + 255) & ~(size_t)255;
    return p;
  };
  float*  m_f  = (float*) wsalloc(sizeof(float)  * MSROWS * DD);   // 16 MB
  __bf16* x_bf = (__bf16*)wsalloc(sizeof(__bf16) * MSROWS * IN_);  //  2 MB
  __bf16* Wm0T = (__bf16*)wsalloc(sizeof(__bf16) * DD * IN_);
  __bf16* WihT = (__bf16*)wsalloc(sizeof(__bf16) * H3 * DD);
  __bf16* WhhT = (__bf16*)wsalloc(sizeof(__bf16) * H3 * HH);
  float*  e_b  = (float*) wsalloc(sizeof(float)  * MSROWS * RR);
  float*  zinv = (float*) wsalloc(sizeof(float)  * MSROWS * RR);
  __bf16* M_bf = (__bf16*)wsalloc(sizeof(__bf16) * MSROWS * DD);   //  8 MB
  float*  gi   = (float*) wsalloc(sizeof(float)  * MSROWS * H3);   // 48 MB

  // 1) precision conversion + weight transposition (N-major bf16 for B-frags)
  k_cvt_bf16  <<<(MSROWS * IN_ + 255) / 256, 256, 0, stream>>>(x, x_bf, MSROWS * IN_);
  k_cvt_bf16_T<<<(IN_ * DD + 255) / 256, 256, 0, stream>>>(W_mlp0, Wm0T, IN_, DD);
  k_cvt_bf16_T<<<(DD * H3 + 255) / 256, 256, 0, stream>>>(W_ih, WihT, DD, H3);
  k_cvt_bf16_T<<<(HH * H3 + 255) / 256, 256, 0, stream>>>(W_hh, WhhT, HH, H3);

  // 2) m = relu(x @ W_mlp0 + b_mlp0)   [8192 x 512, K=128]
  {
    int strips = (MSROWS / 16) * (DD / 64);              // 4096 waves
    k_gemm_wmma<<<strips / 8, 256, 0, stream>>>(x_bf, Wm0T, b_mlp0, m_f,
                                                MSROWS, DD, IN_, 1);
  }
  // 3) e = exp(tanh(m@W1+b1)@W2+b2)
  k_scores<<<MSROWS, 64, 0, stream>>>(m_f, W1, b1, W2, b2, e_b);
  // 4) running 1/Z per (b,r)
  k_zinv<<<1, 256, 0, stream>>>(e_b, zinv);
  // 5) linear-time causal softmax-attention -> M (bf16)
  k_scan<<<BB * 4, 128, 0, stream>>>(m_f, e_b, zinv, M_bf);
  // 6) gi = M @ W_ih + b_i   [8192 x 1536, K=512]
  {
    int strips = (MSROWS / 16) * (H3 / 64);              // 12288 waves
    k_gemm_wmma<<<strips / 8, 256, 0, stream>>>(M_bf, WihT, b_i, gi,
                                                MSROWS, H3, DD, 0);
  }
  // 7) GRU recurrence (persistent, single WGP, W_hh L2-resident)
  k_gru<<<1, 1024, 0, stream>>>(gi, WhhT, b_h, out);
}
